// Model_17136919511634
// MI455X (gfx1250) — matrix operand order repaired and checked by reading, not compiled, not run on hardware
//
#include <hip/hip_runtime.h>
#include <hip/hip_bf16.h>

// ---------------------------------------------------------------------------
// TopK sparse autoencoder forward for MI455X (gfx1250, wave32, WMMA).
//   f    = scatter(relu(top64(x-b_dec @ W_enc^T + b_enc)))      [4096,16384]
//   xhat = f @ W_dec^T + b_dec  (exploiting 64/16384 sparsity)  [4096,1024]
// Encoder GEMM: fp16-in / fp32-accumulate v_wmma_f32_16x16x32_f16.
// Staging: A-tile via GLOBAL_LOAD_ASYNC_TO_LDS_B128 (ASYNCcnt),
//          B-tile via TDM tensor_load_to_lds (TENSORcnt), double-buffered.
// ---------------------------------------------------------------------------

#define BATCH    4096
#define FEATURES 1024
#define HIDDEN   16384
#define TOPK     64

#if __has_builtin(__builtin_amdgcn_global_load_async_to_lds_b128)
#define HAVE_ASYNC 1
#else
#define HAVE_ASYNC 0
#endif

#if __has_include(<hip/amd_detail/amd_gfx1250_TDM.h>) && \
    __has_builtin(__builtin_amdgcn_tensor_load_to_lds)
#define HAVE_TDM6 1   // amdgpu-toolchain 6-arg tensor_load_to_lds
#else
#define HAVE_TDM6 0
#endif

typedef __attribute__((ext_vector_type(16))) _Float16 v16h;
typedef __attribute__((ext_vector_type(8)))  _Float16 v8h;
typedef __attribute__((ext_vector_type(4)))  _Float16 v4h;
typedef __attribute__((ext_vector_type(8)))  float    v8f;
typedef __attribute__((ext_vector_type(4)))  float    v4f;
typedef __attribute__((ext_vector_type(4)))  int      v4i;
typedef __attribute__((ext_vector_type(8)))  int      i32x8;
typedef __attribute__((ext_vector_type(4)))  int      i32x4;
typedef __attribute__((ext_vector_type(4)))  unsigned u32x4;

#if HAVE_ASYNC
// 16B per-lane global->LDS async copy (ASYNCcnt-tracked).
// Builtin prototype (from diagnostics): (AS1 v4i*, AS3 v4i*, imm off, imm cpol)
__device__ __forceinline__ void async_copy16(const _Float16* g, _Float16* l) {
  __builtin_amdgcn_global_load_async_to_lds_b128(
      (__attribute__((address_space(1))) v4i*)(uintptr_t)g,
      (__attribute__((address_space(3))) v4i*)(uint32_t)(uintptr_t)l,
      0, 0);
}
__device__ __forceinline__ void wait_async0() {
#if __has_builtin(__builtin_amdgcn_s_wait_asynccnt)
  __builtin_amdgcn_s_wait_asynccnt(0);
#else
  asm volatile("s_wait_asynccnt 0x0" ::: "memory");
#endif
}
#endif

#if HAVE_TDM6
// TDM load of one B tile: 128 rows x 32 halves, row stride 1024 halves,
// LDS destination padded +4 DWORDs every 16 DWORDs (=> 80B row pitch).
// D# layout per CDNA5 ISA 8.3/8.4; groups 2/3 unused for a 2D tile.
__device__ __forceinline__ void tdm_load_tile_b(const _Float16* gtile,
                                                uint32_t ldsAddrBytes) {
  const uint64_t ga = (uint64_t)(uintptr_t)gtile;
  u32x4 g0;
  g0[0] = 1u;                                   // count=1, user descriptor
  g0[1] = ldsAddrBytes;                         // lds_addr
  g0[2] = (uint32_t)ga;                         // global_addr[31:0]
  g0[3] = (uint32_t)((ga >> 32) & 0x01FFFFFFu)  // global_addr[56:32]
          | (2u << 30);                         // type = 2 ("image")
  i32x8 g1;
  g1[0] = (1 << 16)      // data_size = 2 bytes
        | (1 << 20)      // pad_enable
        | (3 << 22)      // pad_interval: 16 DWORDs (64B = one tile row)
        | (3 << 25);     // pad_amount: 4 DWORDs (16B)
  g1[1] = (int)((FEATURES & 0xFFFF) << 16);     // tensor_dim0 = 1024 (lo16)
  g1[2] = (int)((HIDDEN & 0xFFFF) << 16);       // dim0 hi | tensor_dim1 lo16
  g1[3] = (int)(32 << 16);                      // dim1 hi | tile_dim0 = 32
  g1[4] = 128;                                  // tile_dim1 = 128, tile_dim2=0
  g1[5] = FEATURES;                             // tensor_dim0_stride lo32
  g1[6] = 0;
  g1[7] = 0;
  const i32x4 z4 = {0, 0, 0, 0};
  const i32x8 z8 = {0, 0, 0, 0, 0, 0, 0, 0};
  __builtin_amdgcn_tensor_load_to_lds(g0, g1, z4, z4, z8, 0);
}
__device__ __forceinline__ void wait_tensor0() {
  __builtin_amdgcn_s_wait_tensorcnt(0);
}
#endif

// ---------------------------------------------------------------------------
// Kernel 1: xbar = x - b_dec -> f16 ; W_enc -> f16   (both K-contiguous)
// ---------------------------------------------------------------------------
__global__ __launch_bounds__(256) void sae_prep(
    const float* __restrict__ x, const float* __restrict__ W_enc,
    const float* __restrict__ b_dec,
    _Float16* __restrict__ xh, _Float16* __restrict__ wh) {
  const size_t NX = (size_t)BATCH * FEATURES;
  const size_t NW = (size_t)HIDDEN * FEATURES;
  size_t i4 = ((size_t)blockIdx.x * 256 + threadIdx.x) * 4;
  if (i4 < NX) {
    v4f v  = *(const v4f*)(x + i4);
    v4f bd = *(const v4f*)(b_dec + (i4 & (FEATURES - 1)));
    v4h h;
#pragma unroll
    for (int e = 0; e < 4; ++e) h[e] = (_Float16)(v[e] - bd[e]);
    *(v4h*)(xh + i4) = h;
  } else if (i4 < NX + NW) {
    size_t j = i4 - NX;
    v4f v = *(const v4f*)(W_enc + j);
    v4h h;
#pragma unroll
    for (int e = 0; e < 4; ++e) h[e] = (_Float16)v[e];
    *(v4h*)(wh + j) = h;
  }
}

// ---------------------------------------------------------------------------
// Kernel 2: encoder GEMM. C[m,n] = sum_k A[m,k]*B[n,k] + b_enc[n]
//   A = xbar_f16 [4096,1024], B = Wenc_f16 [16384,1024] (both k-contiguous)
// Block tile 128Mx128N, 8 waves of 32Mx64N, K-step 32, double-buffered LDS.
// ---------------------------------------------------------------------------
#define LDS_PITCH 40   // halves; 80B rows -> conflict-free 16B frag reads

__global__ __launch_bounds__(256) void sae_enc_gemm(
    const _Float16* __restrict__ A, const _Float16* __restrict__ B,
    const float* __restrict__ b_enc, float* __restrict__ out) {
  __shared__ __align__(16) _Float16 As[2][128 * LDS_PITCH];
  __shared__ __align__(16) _Float16 Bs[2][128 * LDS_PITCH];

  const int t    = threadIdx.x;
  const int lane = t & 31;
  const int wave = t >> 5;
  const int m0 = blockIdx.y * 128;
  const int n0 = blockIdx.x * 128;
  const int waveM = (wave & 3) * 32;   // 4 waves tile M
  const int waveN = (wave >> 2) * 64;  // 2 waves tile N

  // staging: each thread owns 32B of the A-tile (and B-tile in fallback)
  const int lrow = t >> 1;
  const int lcol = (t & 1) * 16;
  const _Float16* aSrc = A + (size_t)(m0 + lrow) * FEATURES + lcol;
  const _Float16* bSrc = B + (size_t)(n0 + lrow) * FEATURES + lcol;
  const int ldsOff = lrow * LDS_PITCH + lcol;
  const _Float16* bTile = B + (size_t)n0 * FEATURES;   // TDM tile base (row n0)

  const int lh = lane >> 4;   // half-wave id
  const int ml = lane & 15;   // row (A) / column (B) within 16x16 tile

  v8f acc[2][4] = {};

#if HAVE_ASYNC
  // ---- double-buffered async/TDM global->LDS pipeline: stage tile 0 ----
  async_copy16(aSrc,     &As[0][ldsOff]);
  async_copy16(aSrc + 8, &As[0][ldsOff + 8]);
#if HAVE_TDM6
  if (wave == 0) {
    tdm_load_tile_b(bTile, (uint32_t)(uintptr_t)&Bs[0][0]);
    wait_tensor0();
  }
#else
  async_copy16(bSrc,     &Bs[0][ldsOff]);
  async_copy16(bSrc + 8, &Bs[0][ldsOff + 8]);
#endif
  wait_async0();
  __syncthreads();
#endif

  for (int k0 = 0; k0 < FEATURES; k0 += 32) {
#if HAVE_ASYNC
    const int cur = (k0 >> 5) & 1;
    const int nxt = cur ^ 1;
    const bool more = (k0 + 32) < FEATURES;
    if (more) {  // stream next tile while computing this one
      async_copy16(aSrc + k0 + 32, &As[nxt][ldsOff]);
      async_copy16(aSrc + k0 + 40, &As[nxt][ldsOff + 8]);
#if HAVE_TDM6
      if (wave == 0)
        tdm_load_tile_b(bTile + k0 + 32, (uint32_t)(uintptr_t)&Bs[nxt][0]);
#else
      async_copy16(bSrc + k0 + 32, &Bs[nxt][ldsOff]);
      async_copy16(bSrc + k0 + 40, &Bs[nxt][ldsOff + 8]);
#endif
    }
#else
    const int cur = 0;
    v8h a0 = *(const v8h*)(aSrc + k0);
    v8h a1 = *(const v8h*)(aSrc + k0 + 8);
    v8h b0 = *(const v8h*)(bSrc + k0);
    v8h b1 = *(const v8h*)(bSrc + k0 + 8);
    if (k0 + 32 < FEATURES) {
      __builtin_prefetch(aSrc + k0 + 32, 0, 1);
      __builtin_prefetch(bSrc + k0 + 32, 0, 1);
    }
    __syncthreads();
    *(v8h*)(&As[0][ldsOff])     = a0;
    *(v8h*)(&As[0][ldsOff + 8]) = a1;
    *(v8h*)(&Bs[0][ldsOff])     = b0;
    *(v8h*)(&Bs[0][ldsOff + 8]) = b1;
    __syncthreads();
#endif

    // A fragment: lane ml holds row m; k = 16*lh + {0..15} (contiguous)
    v16h afrag[2];
#pragma unroll
    for (int mi = 0; mi < 2; ++mi) {
      const _Float16* p = &As[cur][(waveM + mi * 16 + ml) * LDS_PITCH];
      v8h lo = *(const v8h*)(p + 16 * lh);
      v8h hi = *(const v8h*)(p + 16 * lh + 8);
#pragma unroll
      for (int e = 0; e < 8; ++e) { afrag[mi][e] = lo[e]; afrag[mi][8 + e] = hi[e]; }
    }
    // B fragment: lane ml holds column n; k = 16*lh + {0..15} (contiguous)
    v16h bfrag[4];
#pragma unroll
    for (int ni = 0; ni < 4; ++ni) {
      const _Float16* p = &Bs[cur][(waveN + ni * 16 + ml) * LDS_PITCH + 16 * lh];
      v8h lo = *(const v8h*)(p);
      v8h hi = *(const v8h*)(p + 8);
#pragma unroll
      for (int e = 0; e < 8; ++e) { bfrag[ni][e] = lo[e]; bfrag[ni][8 + e] = hi[e]; }
    }
#pragma unroll
    for (int mi = 0; mi < 2; ++mi)
#pragma unroll
      for (int ni = 0; ni < 4; ++ni)
        acc[mi][ni] = __builtin_amdgcn_wmma_f32_16x16x32_f16(
            false, afrag[mi], false, bfrag[ni], (short)0, acc[mi][ni], false, false);

#if HAVE_ASYNC
    if (more) {
      asm volatile("s_wait_dscnt 0x0" ::: "memory");  // my LDS reads retired
      wait_async0();                                  // my async writes landed
#if HAVE_TDM6
      if (wave == 0) wait_tensor0();                  // TDM tile landed
#endif
      __syncthreads();                                // everyone else too
    }
#endif
  }

  // epilogue: C element r of v8f -> (M = r + 8*lh, N = ml); add b_enc
#pragma unroll
  for (int mi = 0; mi < 2; ++mi)
#pragma unroll
    for (int ni = 0; ni < 4; ++ni) {
      const int n = n0 + waveN + ni * 16 + ml;
      const float be = b_enc[n];
#pragma unroll
      for (int r = 0; r < 8; ++r) {
        const int m = m0 + waveM + mi * 16 + r + 8 * lh;
        out[(size_t)m * HIDDEN + n] = acc[mi][ni][r] + be;
      }
    }
}

// ---------------------------------------------------------------------------
// Kernel 3: exact per-row top-64 via 3-pass radix select on order keys.
// In-place: row of pre-activations -> sparse relu'd f. Also emits compact
// (idx,val) pairs ordered by index (deterministic decode accumulation).
// ---------------------------------------------------------------------------
__device__ __forceinline__ unsigned order_key(unsigned u) {
  return u ^ ((unsigned)((int)u >> 31) | 0x80000000u);
}

__global__ __launch_bounds__(256) void sae_topk(
    float* __restrict__ f, unsigned* __restrict__ pIdx, float* __restrict__ pVal) {
  __shared__ unsigned hist[2048];
  __shared__ unsigned sBin, sAbove, sTieCnt;
  __shared__ unsigned sTie[256];
  __shared__ unsigned sCnt[256];

  const int row = blockIdx.x;
  const int t = threadIdx.x;
  float* a = f + (size_t)row * HIDDEN;

  unsigned prefix = 0u, pmask = 0u;
  int need = TOPK;
#pragma unroll
  for (int p = 0; p < 3; ++p) {
    const int sh = (p == 0) ? 21 : (p == 1) ? 10 : 0;
    const unsigned nb = (p == 2) ? 1024u : 2048u;
    for (int i = t; i < 2048; i += 256) hist[i] = 0u;
    __syncthreads();
    for (int i = t; i < HIDDEN; i += 256) {
      unsigned key = order_key(__float_as_uint(a[i]));
      if ((key & pmask) == prefix) atomicAdd(&hist[(key >> sh) & (nb - 1u)], 1u);
    }
    __syncthreads();
    if (t == 0) {
      unsigned cum = 0u, b = 0u;
      for (int i = (int)nb - 1; i >= 0; --i) {
        unsigned h = hist[i];
        if (cum + h >= (unsigned)need) { b = (unsigned)i; break; }
        cum += h;
      }
      sBin = b; sAbove = cum;
    }
    __syncthreads();
    need -= (int)sAbove;
    prefix |= sBin << sh;
    pmask |= (nb - 1u) << sh;
    __syncthreads();
  }
  const unsigned T = prefix;  // key of the 64th-largest; take `need` ties

  if (t == 0) sTieCnt = 0u;
  __syncthreads();
  for (int i = t; i < HIDDEN; i += 256) {
    if (order_key(__float_as_uint(a[i])) == T) {
      unsigned s = atomicAdd(&sTieCnt, 1u);
      if (s < 256u) sTie[s] = (unsigned)i;
    }
  }
  __syncthreads();
  const unsigned tieCnt = sTieCnt < 256u ? sTieCnt : 256u;

  // contiguous ownership for deterministic, index-ordered compaction
  const int span = HIDDEN / 256;
  const int base = t * span;
  unsigned myCnt = 0u;
  for (int i = base; i < base + span; ++i) {
    unsigned key = order_key(__float_as_uint(a[i]));
    bool sel = key > T;
    if (key == T) {
      int rank = 0;
      for (unsigned j = 0; j < tieCnt; ++j) rank += (sTie[j] < (unsigned)i) ? 1 : 0;
      sel = rank < need;   // smallest-index ties win: deterministic
    }
    myCnt += sel ? 1u : 0u;
  }
  sCnt[t] = myCnt;
  __syncthreads();
  if (t == 0) {
    unsigned run = 0u;
    for (int i = 0; i < 256; ++i) { unsigned c = sCnt[i]; sCnt[i] = run; run += c; }
  }
  __syncthreads();
  unsigned slot = sCnt[t];
  for (int i = base; i < base + span; ++i) {
    const float v = a[i];
    unsigned key = order_key(__float_as_uint(v));
    bool sel = key > T;
    if (key == T) {
      int rank = 0;
      for (unsigned j = 0; j < tieCnt; ++j) rank += (sTie[j] < (unsigned)i) ? 1 : 0;
      sel = rank < need;
    }
    const float rv = v > 0.f ? v : 0.f;
    a[i] = sel ? rv : 0.f;                       // in-place sparse f
    if (sel) {
      pIdx[(size_t)row * TOPK + slot] = (unsigned)i;
      pVal[(size_t)row * TOPK + slot] = rv;
      ++slot;
    }
  }
}

// ---------------------------------------------------------------------------
// Kernel 4: sparse decode. xhat[b,fe] = b_dec[fe] + sum_j val_j*W_dec[fe,h_j]
// W_dec (64 MB) is L2-resident (192 MB L2) -> gathers hit L2.
// ---------------------------------------------------------------------------
__global__ __launch_bounds__(256) void sae_decode(
    const unsigned* __restrict__ pIdx, const float* __restrict__ pVal,
    const float* __restrict__ W_dec, const float* __restrict__ b_dec,
    float* __restrict__ xhat) {
  __shared__ unsigned sIdx[TOPK];
  __shared__ float sVal[TOPK];
  const int row = blockIdx.x;
  const int t = threadIdx.x;
  if (t < TOPK) {
    sIdx[t] = pIdx[(size_t)row * TOPK + t];
    sVal[t] = pVal[(size_t)row * TOPK + t];
  }
  __syncthreads();
  for (int fe = t; fe < FEATURES; fe += 256) {
    float acc = b_dec[fe];
    const float* wrow = W_dec + (size_t)fe * HIDDEN;
#pragma unroll 8
    for (int j = 0; j < TOPK; ++j) acc += sVal[j] * wrow[sIdx[j]];
    xhat[(size_t)row * FEATURES + fe] = acc;
  }
}

// ---------------------------------------------------------------------------
extern "C" void kernel_launch(void* const* d_in, const int* in_sizes, int n_in,
                              void* d_out, int out_size, void* d_ws, size_t ws_size,
                              hipStream_t stream) {
  const float* x     = (const float*)d_in[0];
  const float* W_enc = (const float*)d_in[1];
  const float* b_enc = (const float*)d_in[2];
  const float* W_dec = (const float*)d_in[3];
  const float* b_dec = (const float*)d_in[4];
  // d_in[5] = activate (64), compiled in as TOPK

  float* f    = (float*)d_out;                       // [4096,16384]
  float* xhat = f + (size_t)BATCH * HIDDEN;          // [4096,1024]

  char* ws = (char*)d_ws;                            // layout (42 MB total):
  _Float16* xh   = (_Float16*)ws;                    //  8 MB xbar f16
  _Float16* wh   = (_Float16*)(ws + ((size_t)8 << 20));   // 32 MB W_enc f16
  unsigned* pIdx = (unsigned*)(ws + ((size_t)40 << 20));  //  1 MB pair idx
  float*    pVal = (float*)   (ws + ((size_t)41 << 20));  //  1 MB pair val

  const size_t total4 = ((size_t)BATCH * FEATURES + (size_t)HIDDEN * FEATURES) / 4;
  const unsigned prepBlocks = (unsigned)((total4 + 255) / 256);
  sae_prep<<<prepBlocks, 256, 0, stream>>>(x, W_enc, b_dec, xh, wh);

  dim3 g(HIDDEN / 128, BATCH / 128);                 // 128 x 32 blocks
  sae_enc_gemm<<<g, 256, 0, stream>>>(xh, wh, b_enc, f);

  sae_topk<<<BATCH, 256, 0, stream>>>(f, pIdx, pVal);

  sae_decode<<<BATCH, 256, 0, stream>>>(pIdx, pVal, W_dec, b_dec, xhat);
}
